// SWTransCrossAttention_67568425501406
// MI455X (gfx1250) — matrix operand order repaired and checked by reading, not compiled, not run on hardware
//
#include <hip/hip_runtime.h>
#include <hip/hip_bf16.h>

typedef __attribute__((ext_vector_type(16))) __bf16 v16bf;
typedef __attribute__((ext_vector_type(8)))  float  v8f;

#define N_TOK   343
#define NPAD    352
#define DIM     192
#define HEADS   6
#define HD      32
#define B_      256
#define BH      (B_*HEADS)            /* 1536 */
#define TOKROWS (B_*N_TOK)            /* 87808 */
#define MT      (TOKROWS/16)          /* 5488 */
#define SLAB    (NPAD*HD)             /* 11264 bf16 per (b,h) */
#define STRQ    ((long)BH*SLAB)       /* per-stream q/k/v elements */
#define STRAO   ((long)TOKROWS*DIM)
#define NN      (N_TOK*N_TOK)         /* 117649 */
#define QSCALE  0.17677669529663687f  /* 32^-0.5 */

__device__ __forceinline__ __bf16 f2bf(float f) { return (__bf16)f; }

union Frag { uint4 q[2]; v16bf v; };

__device__ __forceinline__ v8f wmma_bf16(v16bf a, v16bf b, v8f c) {
  return __builtin_amdgcn_wmma_f32_16x16x32_bf16(false, a, false, b, (short)0, c,
                                                 false, false);
}

// ------------------------------------------------------- x -> bf16 cvt -----
__global__ __launch_bounds__(256)
void cvt_kernel(const float* __restrict__ x0, const float* __restrict__ x1,
                __bf16* __restrict__ Xbf) {
  const float* xs = blockIdx.y ? x1 : x0;
  __bf16* dst = Xbf + (long)blockIdx.y * STRAO;
  long i0 = ((long)blockIdx.x * 256 + threadIdx.x) * 4;
  long stride = (long)gridDim.x * 1024;
  for (long i = i0; i < STRAO; i += stride) {
    float4 v = *(const float4*)(xs + i);
    union { __bf16 h[4]; uint2 u; } t;
    t.h[0] = f2bf(v.x); t.h[1] = f2bf(v.y); t.h[2] = f2bf(v.z); t.h[3] = f2bf(v.w);
    *(uint2*)(dst + i) = t.u;
  }
}

// ---------------------------------------------------------------- prep ------
__global__ __launch_bounds__(256)
void prep_kernel(const float* __restrict__ qkv_w, const float* __restrict__ proj_w,
                 const float* __restrict__ btab, const int* __restrict__ ridx,
                 __bf16* __restrict__ WTqkv, __bf16* __restrict__ WTproj,
                 float* __restrict__ CB) {
  int tid = blockIdx.x * blockDim.x + threadIdx.x;
  int stride = gridDim.x * blockDim.x;
  for (int i = tid; i < DIM * 3 * DIM; i += stride) {      // WTqkv[c][o]
    int c = i / (3 * DIM), o = i % (3 * DIM);
    WTqkv[i] = f2bf(qkv_w[o * DIM + c]);
  }
  for (int i = tid; i < DIM * DIM; i += stride) {          // WTproj[c][o]
    int c = i / DIM, o = i % DIM;
    WTproj[i] = f2bf(proj_w[o * DIM + c]);
  }
  for (int i = tid; i < NN; i += stride) {                 // CB[h][n][m]
    int idx = ridx[i];
#pragma unroll
    for (int h = 0; h < HEADS; ++h)
      CB[h * NN + i] = btab[idx * HEADS + h];
  }
}

// ----------------------------------------------- zero q/k/v pad regions ----
__global__ __launch_bounds__(256)
void padzero_kernel(__bf16* __restrict__ Q, __bf16* __restrict__ KT,
                    __bf16* __restrict__ V) {
  long base = (long)blockIdx.x * SLAB;     // slab = s*BH + bh (STRQ = BH*SLAB)
  for (int i = threadIdx.x; i < 288; i += 256) {
    int rr = i >> 5, d = i & 31;           // pad rows 343..351 of Q,V
    Q[base + (N_TOK + rr) * HD + d] = f2bf(0.f);
    V[base + (N_TOK + rr) * HD + d] = f2bf(0.f);
    int dd = i / 9, cc = i - dd * 9;       // pad cols 343..351 of K^T
    KT[base + dd * NPAD + N_TOK + cc] = f2bf(0.f);
  }
}

// ------------------------------------------------------------ qkv gemm -----
// Each wave: 16 rows x 64 cols (4 N-tiles), K = 192 in 6 chunks of 32.
__global__ __launch_bounds__(256)
void qkv_gemm_kernel(const __bf16* __restrict__ Xbf, const __bf16* __restrict__ WT,
                     const float* __restrict__ qb, __bf16* __restrict__ Qb,
                     __bf16* __restrict__ KTb, __bf16* __restrict__ Vb) {
  const int lane = threadIdx.x & 31, wave = threadIdx.x >> 5;
  const int ln = lane & 15, hl = lane >> 4;
  long tile = (long)blockIdx.x * 8 + wave;          // [0, MT*9)
  int mtile = (int)(tile % MT), ng = (int)(tile / MT);
  const __bf16* Xs = Xbf + (long)blockIdx.y * STRAO;
  __bf16* Qs  = Qb  + (long)blockIdx.y * STRQ;
  __bf16* KTs = KTb + (long)blockIdx.y * STRQ;
  __bf16* Vs  = Vb  + (long)blockIdx.y * STRQ;

  const int row = mtile * 16 + ln;
  v8f acc[4] = {{}, {}, {}, {}};
#pragma unroll
  for (int kc = 0; kc < 6; ++kc) {
    const int cb = kc * 32;
    const __bf16* ap = Xs + (long)row * DIM + cb + 8 * hl;
    Frag A; A.q[0] = *(const uint4*)(ap); A.q[1] = *(const uint4*)(ap + 16);
    const __bf16* bp = WT + (long)(cb + 8 * hl) * (3 * DIM) + ng * 64 + ln;
#pragma unroll
    for (int t = 0; t < 4; ++t) {
      Frag B;
#pragma unroll
      for (int s = 0; s < 8; ++s) {
        B.v[s]     = bp[s * (3 * DIM) + t * 16];
        B.v[s + 8] = bp[(s + 16) * (3 * DIM) + t * 16];
      }
      acc[t] = wmma_bf16(A.v, B.v, acc[t]);
    }
  }
  int brow[8], nrow[8];
#pragma unroll
  for (int r = 0; r < 8; ++r) {
    int t = mtile * 16 + r + 8 * hl;
    brow[r] = t / N_TOK; nrow[r] = t - brow[r] * N_TOK;
  }
#pragma unroll
  for (int t = 0; t < 4; ++t) {
    const int o = (ng * 4 + t) * 16 + ln;
    const float bias = qb[o];
    if (o < DIM) {                                 // Q (pre-scaled)
      int h = o >> 5, d = o & 31;
#pragma unroll
      for (int r = 0; r < 8; ++r)
        Qs[((long)(brow[r] * HEADS + h) * NPAD + nrow[r]) * HD + d] =
            f2bf((acc[t][r] + bias) * QSCALE);
    } else if (o < 2 * DIM) {                      // K -> transposed [bh][d][n]
      int oo = o - DIM; int h = oo >> 5, d = oo & 31;
#pragma unroll
      for (int r = 0; r < 8; ++r)
        KTs[((long)(brow[r] * HEADS + h) * HD + d) * NPAD + nrow[r]] =
            f2bf(acc[t][r] + bias);
    } else {                                       // V [bh][n][d]
      int oo = o - 2 * DIM; int h = oo >> 5, d = oo & 31;
#pragma unroll
      for (int r = 0; r < 8; ++r)
        Vs[((long)(brow[r] * HEADS + h) * NPAD + nrow[r]) * HD + d] =
            f2bf(acc[t][r] + bias);
    }
  }
}

// ----------------------------------------------------------- attention -----
#define KTP 360   /* padded LDS row stride (bf16) for K^T rows */
#define VP  40    /* padded LDS row stride for V rows */
#define PP  40    /* padded LDS row stride for P transpose buffer */

__global__ __launch_bounds__(256)
void attn_kernel(const __bf16* __restrict__ Qb, const __bf16* __restrict__ KTb,
                 const __bf16* __restrict__ Vb, const float* __restrict__ CB,
                 const float* __restrict__ mask, __bf16* __restrict__ AOb) {
  __shared__ __bf16 sKT[32 * KTP];
  __shared__ __bf16 sV[NPAD * VP];
  __shared__ __bf16 sP[8][16 * PP];

  const int bh = blockIdx.x;          // 0..1535
  const int qs = blockIdx.y;          // query stream; kv from other stream
  const int b = bh / HEADS, h = bh - b * HEADS;
  const int w = b & 63;
  const __bf16* Qp  = Qb  + (long)qs * STRQ + (long)bh * SLAB;
  const __bf16* KTp = KTb + (long)(1 - qs) * STRQ + (long)bh * SLAB;
  const __bf16* Vp  = Vb  + (long)(1 - qs) * STRQ + (long)bh * SLAB;
  __bf16* AOs = AOb + (long)qs * STRAO;

  // Stage K^T (32x352) and V (352x32) into padded LDS, dword-wise.
  for (int i = threadIdx.x; i < SLAB / 2; i += 256) {
    { int d = i / 176, j = i - d * 176;        // K^T: 176 dwords per row
      ((unsigned*)sKT)[d * (KTP / 2) + j] = ((const unsigned*)KTp)[i]; }
    { int rrow = i / 16, j = i - rrow * 16;    // V: 16 dwords per row
      ((unsigned*)sV)[rrow * (VP / 2) + j] = ((const unsigned*)Vp)[i]; }
  }
  __syncthreads();

  const int lane = threadIdx.x & 31, wave = threadIdx.x >> 5;
  const int ln = lane & 15, hl = lane >> 4;
  const float* CBh = CB + h * NN;
  const float* Mw  = mask + (long)w * NN;
  __bf16* pw = sP[wave];

  for (int rt = wave; rt < 22; rt += 8) {
    const __bf16* qp = Qp + (rt * 16 + ln) * HD;
    Frag QA;
    QA.q[0] = *(const uint4*)(qp + 8 * hl);
    QA.q[1] = *(const uint4*)(qp + 16 + 8 * hl);

    float mrow[8], lrow[8];
    v8f O0 = {}, O1 = {};
#pragma unroll
    for (int r = 0; r < 8; ++r) { mrow[r] = -3e38f; lrow[r] = 0.f; }

    for (int jc = 0; jc < 11; ++jc) {
      // --- S = Q K^T : two 16x16 tiles, K = hd = 32 in one WMMA each
      const __bf16* kp = sKT + (8 * hl) * KTP + jc * 32 + ln;
      Frag B0, B1;
#pragma unroll
      for (int s = 0; s < 8; ++s) {
        B0.v[s] = kp[s * KTP];      B0.v[s + 8] = kp[(s + 16) * KTP];
        B1.v[s] = kp[s * KTP + 16]; B1.v[s + 8] = kp[(s + 16) * KTP + 16];
      }
      v8f S0 = {}, S1 = {};
      S0 = wmma_bf16(QA.v, B0.v, S0);
      S1 = wmma_bf16(QA.v, B1.v, S1);

      // --- add rel-bias + window mask, -inf on padded columns
      const int c0 = jc * 32 + ln, c1 = c0 + 16;
      const int c0c = c0 > 342 ? 342 : c0;
      const int c1c = c1 > 342 ? 342 : c1;
      const bool ok0 = c0 < N_TOK, ok1 = c1 < N_TOK;
#pragma unroll
      for (int r = 0; r < 8; ++r) {
        int n = rt * 16 + r + 8 * hl; int nc = n > 342 ? 342 : n;
        long base = (long)nc * N_TOK;
        float s0 = ok0 ? S0[r] + CBh[base + c0c] + Mw[base + c0c] : -3e38f;
        float s1 = ok1 ? S1[r] + CBh[base + c1c] + Mw[base + c1c] : -3e38f;
        S0[r] = s0; S1[r] = s1;
      }

      // --- online softmax (rows live across 16-lane halves)
#pragma unroll
      for (int r = 0; r < 8; ++r) {
        float tmax = fmaxf(S0[r], S1[r]);
        tmax = fmaxf(tmax, __shfl_xor(tmax, 1));
        tmax = fmaxf(tmax, __shfl_xor(tmax, 2));
        tmax = fmaxf(tmax, __shfl_xor(tmax, 4));
        tmax = fmaxf(tmax, __shfl_xor(tmax, 8));
        float newm = fmaxf(mrow[r], tmax);
        float sc = __expf(mrow[r] - newm);
        mrow[r] = newm;
        lrow[r] *= sc;
        O0[r] *= sc; O1[r] *= sc;
        float p0 = __expf(S0[r] - newm);
        float p1 = __expf(S1[r] - newm);
        float ps = p0 + p1;
        ps += __shfl_xor(ps, 1);
        ps += __shfl_xor(ps, 2);
        ps += __shfl_xor(ps, 4);
        ps += __shfl_xor(ps, 8);
        lrow[r] += ps;
        S0[r] = p0; S1[r] = p1;
      }

      // --- transpose P (C-layout -> A-layout) through per-wave LDS
#pragma unroll
      for (int r = 0; r < 8; ++r) {
        int rr = r + 8 * hl;
        pw[rr * PP + ln]      = f2bf(S0[r]);
        pw[rr * PP + 16 + ln] = f2bf(S1[r]);
      }
      const __bf16* pp = pw + ln * PP;
      Frag PA;
      PA.q[0] = *(const uint4*)(pp + 8 * hl);
      PA.q[1] = *(const uint4*)(pp + 16 + 8 * hl);

      // --- O += P V : V B-fragments, lane = output column d, elements = tokens
      const __bf16* vp = sV + (jc * 32 + 8 * hl) * VP + ln;
      Frag V0, V1;
#pragma unroll
      for (int s = 0; s < 8; ++s) {
        V0.v[s] = vp[s * VP];      V0.v[s + 8] = vp[(s + 16) * VP];
        V1.v[s] = vp[s * VP + 16]; V1.v[s + 8] = vp[(s + 16) * VP + 16];
      }
      O0 = wmma_bf16(PA.v, V0.v, O0);
      O1 = wmma_bf16(PA.v, V1.v, O1);
    }

    // --- normalize and emit bf16 attention output [b][n][h*32+d]
#pragma unroll
    for (int r = 0; r < 8; ++r) {
      int n = rt * 16 + r + 8 * hl;
      if (n < N_TOK) {
        float inv = 1.0f / lrow[r];
        long rowoff = ((long)b * N_TOK + n) * DIM + h * HD;
        AOs[rowoff + ln]      = f2bf(O0[r] * inv);
        AOs[rowoff + 16 + ln] = f2bf(O1[r] * inv);
      }
    }
  }
}

// ----------------------------------------------------------- proj gemm -----
// Each wave: 16 rows x 64 cols (4 N-tiles), 3 groups cover DIM = 192.
__global__ __launch_bounds__(256)
void proj_gemm_kernel(const __bf16* __restrict__ AOb, const __bf16* __restrict__ WT,
                      const float* __restrict__ pb, float* __restrict__ OUT) {
  const int lane = threadIdx.x & 31, wave = threadIdx.x >> 5;
  const int ln = lane & 15, hl = lane >> 4;
  long tile = (long)blockIdx.x * 8 + wave;          // [0, MT*3)
  int mtile = (int)(tile % MT), ng = (int)(tile / MT);
  const __bf16* As = AOb + (long)blockIdx.y * STRAO;
  float* Os = OUT + (long)blockIdx.y * STRAO;

  const int row = mtile * 16 + ln;
  v8f acc[4] = {{}, {}, {}, {}};
#pragma unroll
  for (int kc = 0; kc < 6; ++kc) {
    const int cb = kc * 32;
    const __bf16* ap = As + (long)row * DIM + cb + 8 * hl;
    Frag A; A.q[0] = *(const uint4*)(ap); A.q[1] = *(const uint4*)(ap + 16);
    const __bf16* bp = WT + (long)(cb + 8 * hl) * DIM + ng * 64 + ln;
#pragma unroll
    for (int t = 0; t < 4; ++t) {
      Frag B;
#pragma unroll
      for (int s = 0; s < 8; ++s) {
        B.v[s]     = bp[s * DIM + t * 16];
        B.v[s + 8] = bp[(s + 16) * DIM + t * 16];
      }
      acc[t] = wmma_bf16(A.v, B.v, acc[t]);
    }
  }
#pragma unroll
  for (int t = 0; t < 4; ++t) {
    const int col = (ng * 4 + t) * 16 + ln;
    const float bias = pb[col];
#pragma unroll
    for (int r = 0; r < 8; ++r) {
      int trow = mtile * 16 + r + 8 * hl;
      Os[(long)trow * DIM + col] = acc[t][r] + bias;
    }
  }
}

// ------------------------------------------------------------- launcher ----
extern "C" void kernel_launch(void* const* d_in, const int* in_sizes, int n_in,
                              void* d_out, int out_size, void* d_ws, size_t ws_size,
                              hipStream_t stream) {
  const float* x      = (const float*)d_in[0];
  const float* x1     = (const float*)d_in[1];
  const float* mask   = (const float*)d_in[2];
  const float* qkv_w  = (const float*)d_in[3];
  const float* qkv_b  = (const float*)d_in[4];
  const float* proj_w = (const float*)d_in[5];
  const float* proj_b = (const float*)d_in[6];
  const float* btab   = (const float*)d_in[7];
  const int*   ridx   = (const int*)d_in[8];

  char* ws = (char*)d_ws;
  // byte offsets (256-aligned)
  __bf16* WTqkv  = (__bf16*)(ws + 0);                 //   221,184 B
  __bf16* WTproj = (__bf16*)(ws + 221184);            //    73,728 B
  float*  CB     = (float*)(ws + 294912);             // 2,823,576 B
  __bf16* Q      = (__bf16*)(ws + 3118592);           // 69,206,016 B
  __bf16* KT     = (__bf16*)(ws + 72324608);          // 69,206,016 B
  __bf16* V      = (__bf16*)(ws + 141530624);         // 69,206,016 B
  __bf16* AO     = (__bf16*)(ws + 210736640);         // 67,436,544 B
  __bf16* Xbf    = (__bf16*)(ws + 278173184);         // 67,436,544 B
  float* out = (float*)d_out;

  padzero_kernel<<<dim3(2 * BH), dim3(256), 0, stream>>>(Q, KT, V);
  cvt_kernel<<<dim3(4096, 2), dim3(256), 0, stream>>>(x, x1, Xbf);
  prep_kernel<<<dim3(720), dim3(256), 0, stream>>>(qkv_w, proj_w, btab, ridx,
                                                   WTqkv, WTproj, CB);
  qkv_gemm_kernel<<<dim3((MT * 9) / 8, 2), dim3(256), 0, stream>>>(
      Xbf, WTqkv, qkv_b, Q, KT, V);
  attn_kernel<<<dim3(BH, 2), dim3(256), 0, stream>>>(Q, KT, V, CB, mask, AO);
  proj_gemm_kernel<<<dim3((MT * 3) / 8, 2), dim3(256), 0, stream>>>(
      AO, WTproj, proj_b, out);
}
